// Model_77730318123211
// MI455X (gfx1250) — matrix operand order repaired and checked by reading, not compiled, not run on hardware
//
#include <hip/hip_runtime.h>

typedef _Float16 h16;
typedef __attribute__((ext_vector_type(16))) _Float16 v16h;
typedef __attribute__((ext_vector_type(8)))  _Float16 v8h;
typedef __attribute__((ext_vector_type(8)))  float    v8f;

#define B_   16
#define T_   24
#define N_   2000
#define NP_  2048     // N padded to tile multiple
#define H_   64
#define E_   3
#define OUT_ 24
#define CDIV(a,b) (((a)+(b)-1)/(b))

// ------------------------------------------------------------------
// WMMA GEMM, fully padded operands: M multiple of 128, Nc multiple of
// 64, K multiple of 32; lda=K, ldb=ldc=Nc. Block = 256 thr (8 waves),
// 128x64 tile; each wave owns 1 m-frag x 4 n-frags -> 4 WMMA per
// iteration off 10 b128 LDS loads + 2 barriers.
// A/B fragments: contiguous K, k = 16*(lane/16) + s, same for both (not ISA table order).
// ------------------------------------------------------------------
template<bool HALF_OUT>
__global__ __launch_bounds__(256) void k_gemm(
    const h16* __restrict__ A, const h16* __restrict__ Bm,
    void* __restrict__ Cv, int M, int Nc, int K)
{
  __shared__ alignas(16) h16 lA [128][40];  // A tile 128x32 (+pad)
  __shared__ alignas(16) h16 lBt[64][40];   // B tile transposed: [n][k]
  const int tid  = threadIdx.x;
  const int wave = tid >> 5, lane = tid & 31;
  const int m0 = blockIdx.y * 128, n0 = blockIdx.x * 64;
  const int am    = wave * 16 + (lane & 15);    // A row for this lane
  const int abase = (lane >> 4) * 16;           // K sub-base {0,16}
  const int bnl   = lane & 15;                  // col within n-frag
  const int kb0   = (lane >> 4) * 16;           // B K-base {0,16}

  // tile-load coordinates: 16B vectors per thread
  const int ar = tid >> 2, ac = (tid & 3) * 8;  // A: rows ar, ar+64
  const int br = tid >> 3, bc = (tid & 7) * 8;  // B: 32 rows x 64 cols
  const h16* Ap0 = A  + (size_t)(m0 + ar) * K + ac;
  const h16* Ap1 = A  + (size_t)(m0 + ar + 64) * K + ac;
  const h16* Bp  = Bm + (size_t)br * Nc + n0 + bc;

  v8f acc0 = {}, acc1 = {}, acc2 = {}, acc3 = {};

  for (int kk = 0; kk < K; kk += 32) {
    v8h av0 = *(const v8h*)(Ap0 + kk);
    v8h av1 = *(const v8h*)(Ap1 + kk);
    v8h bv  = *(const v8h*)(Bp + (size_t)kk * Nc);
    if (kk + 32 < K) {                          // CDNA5 global_prefetch_b8
      __builtin_prefetch(Ap0 + kk + 32, 0, 1);
      __builtin_prefetch(Ap1 + kk + 32, 0, 1);
    }
    *(v8h*)&lA[ar][ac]      = av0;
    *(v8h*)&lA[ar + 64][ac] = av1;
#pragma unroll
    for (int j = 0; j < 8; ++j) lBt[bc + j][br] = bv[j];
    __syncthreads();

    v16h va, vb0, vb1, vb2, vb3;
    v8h alo = *(const v8h*)&lA[am][abase];        // K = base+0..7
    v8h ahi = *(const v8h*)&lA[am][abase + 8];    // K = base+8..15
    v8h b0a = *(const v8h*)&lBt[bnl     ][kb0];
    v8h b0b = *(const v8h*)&lBt[bnl     ][kb0 + 8];
    v8h b1a = *(const v8h*)&lBt[bnl + 16][kb0];
    v8h b1b = *(const v8h*)&lBt[bnl + 16][kb0 + 8];
    v8h b2a = *(const v8h*)&lBt[bnl + 32][kb0];
    v8h b2b = *(const v8h*)&lBt[bnl + 32][kb0 + 8];
    v8h b3a = *(const v8h*)&lBt[bnl + 48][kb0];
    v8h b3b = *(const v8h*)&lBt[bnl + 48][kb0 + 8];
#pragma unroll
    for (int j = 0; j < 8; ++j) {
      va[j]  = alo[j]; va[8 + j]  = ahi[j];
      vb0[j] = b0a[j]; vb0[8 + j] = b0b[j];
      vb1[j] = b1a[j]; vb1[8 + j] = b1b[j];
      vb2[j] = b2a[j]; vb2[8 + j] = b2b[j];
      vb3[j] = b3a[j]; vb3[8 + j] = b3b[j];
    }
    acc0 = __builtin_amdgcn_wmma_f32_16x16x32_f16(false, va, false, vb0,
                                                  (short)0, acc0, false, false);
    acc1 = __builtin_amdgcn_wmma_f32_16x16x32_f16(false, va, false, vb1,
                                                  (short)0, acc1, false, false);
    acc2 = __builtin_amdgcn_wmma_f32_16x16x32_f16(false, va, false, vb2,
                                                  (short)0, acc2, false, false);
    acc3 = __builtin_amdgcn_wmma_f32_16x16x32_f16(false, va, false, vb3,
                                                  (short)0, acc3, false, false);
    __syncthreads();
  }

  // C/D layout: VGPR v, lane l -> row = v + 8*(l>=16), col = l%16
  const int rbase = m0 + wave * 16 + ((lane >> 4) << 3);
  const int c0 = n0 + bnl;
#pragma unroll
  for (int v = 0; v < 8; ++v) {
    size_t r = (size_t)(rbase + v) * Nc;
    if (HALF_OUT) {
      h16* C = (h16*)Cv;
      C[r + c0]      = (h16)acc0[v];
      C[r + c0 + 16] = (h16)acc1[v];
      C[r + c0 + 32] = (h16)acc2[v];
      C[r + c0 + 48] = (h16)acc3[v];
    } else {
      float* C = (float*)Cv;
      C[r + c0]      = acc0[v];
      C[r + c0 + 16] = acc1[v];
      C[r + c0 + 32] = acc2[v];
      C[r + c0 + 48] = acc3[v];
    }
  }
}

// ------------------------- small kernels --------------------------
__global__ void k_cvt(const float* __restrict__ s, h16* __restrict__ d, int n) {
  int i = blockIdx.x * blockDim.x + threadIdx.x;
  if (i < n) d[i] = (h16)s[i];
}

__global__ void k_zero(float* p, int n) {
  int i = blockIdx.x * blockDim.x + threadIdx.x;
  if (i < n) p[i] = 0.f;
}

__global__ void k_zero16(h16* p, size_t n) {
  size_t i = (size_t)blockIdx.x * blockDim.x + threadIdx.x;
  size_t st = (size_t)gridDim.x * blockDim.x;
  for (; i < n; i += st) p[i] = (h16)0.f;
}

// per-row sums of adj (all, binarized no-self, weighted no-self)
__global__ void k_rowstats(const float* __restrict__ adj,
                           float* dall, float* dbin, float* dns) {
  __shared__ float sa[256], sb[256], sn[256];
  int n = blockIdx.x, tid = threadIdx.x;
  float a = 0.f, b = 0.f, c = 0.f;
  for (int m = tid; m < N_; m += 256) {
    float v = adj[(size_t)n * N_ + m];
    a += v;
    float vns = (m == n) ? 0.f : v;
    c += vns;
    b += (vns > 0.f) ? 1.f : 0.f;
  }
  sa[tid] = a; sb[tid] = b; sn[tid] = c; __syncthreads();
  for (int s = 128; s > 0; s >>= 1) {
    if (tid < s) { sa[tid] += sa[tid+s]; sb[tid] += sb[tid+s]; sn[tid] += sn[tid+s]; }
    __syncthreads();
  }
  if (tid == 0) { dall[n] = sa[0]; dbin[n] = sb[0]; dns[n] = sn[0]; }
}

__global__ void k_colsum(const float* __restrict__ adj, float* dcol) {
  int j = blockIdx.x * blockDim.x + threadIdx.x;
  if (j >= N_) return;
  float s = 0.f;
  for (int m = 0; m < N_; ++m) s += adj[(size_t)m * N_ + j];
  dcol[j] = s;
}

// Build A_q, A_h, A_mean, A_wmean(==A_diff) in f16, padded stride NP_
__global__ void k_build_adj(const float* __restrict__ adj,
                            const float* dall, const float* dcol,
                            const float* dbin, const float* dns,
                            h16* Aq, h16* Ah, h16* Am, h16* Aw) {
  size_t idx = (size_t)blockIdx.x * 256 + threadIdx.x;
  if (idx >= (size_t)N_ * N_) return;
  int n = (int)(idx / N_), m = (int)(idx % N_);
  float a  = adj[idx];
  float at = adj[(size_t)m * N_ + n];
  float ans = (m == n) ? 0.f : a;
  float dq = dall[n]; dq = (dq == 0.f) ? 1.f : dq;
  float dh = dcol[n]; dh = (dh == 0.f) ? 1.f : dh;
  float db = fmaxf(dbin[n], 1.f);
  float dw = dns[n];  dw = (dw == 0.f) ? 1.f : dw;
  size_t o = (size_t)n * NP_ + m;
  Aq[o] = (h16)(a  / dq);
  Ah[o] = (h16)(at / dh);
  Am[o] = (h16)(((ans > 0.f) ? 1.f : 0.f) / db);
  Aw[o] = (h16)(ans / dw);
}

// Xn[n][b*T+t] = x_enc[b,t,n]  (f16)
__global__ void k_xn(const float* __restrict__ xe, h16* __restrict__ Xn) {
  int i = blockIdx.x * blockDim.x + threadIdx.x;
  if (i >= N_ * B_ * T_) return;
  int n = i / (B_ * T_), rem = i % (B_ * T_);
  int b = rem / T_, t = rem % T_;
  Xn[i] = (h16)xe[((size_t)b * T_ + t) * N_ + n];
}

// MoE gate: relu(x@W1+b1)@W2+b2, top-2 of 3, softmax
__global__ void k_gate(const float* __restrict__ xe,
                       const float* gw1, const float* gb1,
                       const float* gw2, const float* gb2,
                       float* __restrict__ gate) {
  int idx = blockIdx.x * blockDim.x + threadIdx.x;
  if (idx >= B_ * N_) return;
  int b = idx / N_, n = idx % N_;
  float xr[T_];
  for (int t = 0; t < T_; ++t) xr[t] = xe[((size_t)b * T_ + t) * N_ + n];
  float lg[E_];
  for (int e = 0; e < E_; ++e) lg[e] = gb2[e];
  for (int j = 0; j < H_; ++j) {
    float g = gb1[j];
    for (int t = 0; t < T_; ++t) g += xr[t] * gw1[t * H_ + j];
    g = fmaxf(g, 0.f);
    for (int e = 0; e < E_; ++e) lg[e] += g * gw2[j * E_ + e];
  }
  int i1 = 0;
  for (int e = 1; e < E_; ++e) if (lg[e] > lg[i1]) i1 = e;
  int i2 = -1;
  for (int e = 0; e < E_; ++e) {
    if (e == i1) continue;
    if (i2 < 0 || lg[e] > lg[i2]) i2 = e;
  }
  float e1 = 1.f, e2 = expf(lg[i2] - lg[i1]);
  float s = e1 + e2;
  float o[E_] = {0.f, 0.f, 0.f};
  o[i1] = e1 / s; o[i2] = e2 / s;
  for (int e = 0; e < E_; ++e) gate[(size_t)idx * E_ + e] = o[e];
}

// temporal mix + gated combine + imputation; ximp layout [t][b][n]
__global__ void k_combine(const float* __restrict__ Ym, const float* __restrict__ Yw,
                          const float* ewm, const float* ebm,
                          const float* eww, const float* ebw,
                          const float* ewd, const float* ebd,
                          const float* __restrict__ gate,
                          const float* __restrict__ xe,
                          float* __restrict__ ximp) {
  int idx = blockIdx.x * blockDim.x + threadIdx.x;
  if (idx >= B_ * N_) return;
  int b = idx / N_, n = idx % N_;
  const float* y0 = Ym + (size_t)n * (B_ * T_) + b * T_;
  const float* y1 = Yw + (size_t)n * (B_ * T_) + b * T_;   // A_diff == A_wmean
  float g0 = gate[(size_t)idx * E_ + 0];
  float g1 = gate[(size_t)idx * E_ + 1];
  float g2 = gate[(size_t)idx * E_ + 2];
  for (int t = 0; t < T_; ++t) {
    float o0 = ebm[t], o1 = ebw[t], o2 = ebd[t];
    for (int s = 0; s < T_; ++s) {
      float a = y0[s], c = y1[s];
      o0 += a * ewm[s * T_ + t];
      o1 += c * eww[s * T_ + t];
      o2 += c * ewd[s * T_ + t];
    }
    float comb = g0 * o0 + g1 * o1 + g2 * o2;
    float xv = xe[((size_t)b * T_ + t) * N_ + n];
    ximp[(size_t)t * B_ * N_ + b * N_ + n] = (xv == 0.f) ? comb : xv;
  }
}

// Z[n][b*f + c] = concat(inp, (r?)*h)  -> f16 diffusion operand, stride colsP
__global__ void k_pack_xh(const float* __restrict__ inp, int fin,
                          const float* __restrict__ h, const float* __restrict__ r,
                          h16* __restrict__ Z, int colsP) {
  int f = fin + H_;
  int total = N_ * B_ * f;
  int idx = blockIdx.x * blockDim.x + threadIdx.x;
  if (idx >= total) return;
  int n = idx / (B_ * f), rem = idx % (B_ * f);
  int b = rem / f, c = rem % f;
  int row = b * N_ + n;
  float v;
  if (c < fin) v = (fin == 1) ? inp[row] : inp[(size_t)row * fin + c];
  else {
    int j = c - fin;
    v = h[(size_t)row * H_ + j];
    if (r) v *= r[(size_t)row * 2 * H_ + j];
  }
  Z[(size_t)n * colsP + b * f + c] = (h16)v;
}

// G[b*N+n][5f] = [Z, Aq x, Aq^2 x, Ah x, Ah^2 x]; chunk stride colsP, G stride fcatP
__global__ void k_pack_cat(const h16* __restrict__ Z,
                           const h16* __restrict__ Y1, const h16* __restrict__ Y2,
                           const h16* __restrict__ Y3, const h16* __restrict__ Y4,
                           h16* __restrict__ G, int f, int colsP, int fcatP) {
  int fcat = 5 * f;
  int total = B_ * N_ * fcat;
  int idx = blockIdx.x * blockDim.x + threadIdx.x;
  if (idx >= total) return;
  int row = idx / fcat, k = idx % fcat;
  int b = row / N_, n = row % N_;
  int ch = k / f, c = k % f;
  const h16* src = (ch == 0) ? Z : (ch == 1) ? Y1 : (ch == 2) ? Y2 : (ch == 3) ? Y3 : Y4;
  G[(size_t)row * fcatP + k] = src[(size_t)n * colsP + b * f + c];
}

__global__ void k_bias_sigmoid(float* y, const float* b, int total, int cols) {
  int i = blockIdx.x * blockDim.x + threadIdx.x;
  if (i >= total) return;
  float v = y[i] + b[i % cols];
  y[i] = 1.f / (1.f + expf(-v));
}

__global__ void k_bias_tanh(float* y, const float* b, int total, int cols) {
  int i = blockIdx.x * blockDim.x + threadIdx.x;
  if (i >= total) return;
  y[i] = tanhf(y[i] + b[i % cols]);
}

__global__ void k_gru(float* __restrict__ h, const float* __restrict__ ru,
                      const float* __restrict__ c) {
  int i = blockIdx.x * blockDim.x + threadIdx.x;
  if (i >= B_ * N_ * H_) return;
  int row = i / H_;
  int j = i % H_;
  float u = ru[(size_t)row * 2 * H_ + H_ + j];
  h[i] = u * h[i] + (1.f - u) * c[i];
}

__global__ void k_fc(const float* __restrict__ h1, const float* fcw, const float* fcb,
                     float* __restrict__ out, float* __restrict__ dec_in, int t) {
  int idx = blockIdx.x * blockDim.x + threadIdx.x;
  if (idx >= B_ * N_) return;
  int b = idx / N_, n = idx % N_;
  float s = fcb[0];
  for (int j = 0; j < H_; ++j) s += h1[(size_t)idx * H_ + j] * fcw[j];
  out[((size_t)b * OUT_ + t) * N_ + n] = s;
  dec_in[idx] = s;
}

// ------------------------- host orchestration --------------------------
static inline void gemm16(const h16* A, const h16* B, h16* C,
                          int M, int Nc, int K, hipStream_t s) {
  dim3 g(Nc / 64, M / 128);
  k_gemm<true><<<g, 256, 0, s>>>(A, B, (void*)C, M, Nc, K);
}
static inline void gemm32(const h16* A, const h16* B, float* C,
                          int M, int Nc, int K, hipStream_t s) {
  dim3 g(Nc / 64, M / 128);
  k_gemm<false><<<g, 256, 0, s>>>(A, B, (void*)C, M, Nc, K);
}

extern "C" void kernel_launch(void* const* d_in, const int* in_sizes, int n_in,
                              void* d_out, int out_size, void* d_ws, size_t ws_size,
                              hipStream_t stream) {
  const float* adj    = (const float*)d_in[0];
  const float* x_enc  = (const float*)d_in[1];
  const float* gw1    = (const float*)d_in[3];
  const float* gb1    = (const float*)d_in[4];
  const float* gw2    = (const float*)d_in[5];
  const float* gb2    = (const float*)d_in[6];
  const float* ewm    = (const float*)d_in[7];
  const float* ebm    = (const float*)d_in[8];
  const float* eww    = (const float*)d_in[9];
  const float* ebw    = (const float*)d_in[10];
  const float* ewd    = (const float*)d_in[11];
  const float* ebd    = (const float*)d_in[12];
  const float* Wg_f32[4] = { (const float*)d_in[13], (const float*)d_in[17],
                             (const float*)d_in[21], (const float*)d_in[25] };
  const float* bg_f32[4] = { (const float*)d_in[14], (const float*)d_in[18],
                             (const float*)d_in[22], (const float*)d_in[26] };
  const float* Wc_f32[4] = { (const float*)d_in[15], (const float*)d_in[19],
                             (const float*)d_in[23], (const float*)d_in[27] };
  const float* bc_f32[4] = { (const float*)d_in[16], (const float*)d_in[20],
                             (const float*)d_in[24], (const float*)d_in[28] };
  const float* fcw = (const float*)d_in[29];
  const float* fcb = (const float*)d_in[30];

  const int fcat0  = 5 * (1 + H_);     // 325
  const int fcat0P = 352;              // padded to x32
  const int fcat1  = 5 * (2 * H_);     // 640 (already x32)
  const int colsP0 = 1088;             // B_*65=1040 -> x64
  const int colsP1 = 2048;             // B_*128 (already x64)
  const size_t NPP = (size_t)NP_ * NP_;

  // bump allocator over d_ws
  char* ws = (char*)d_ws;
  size_t off = 0;
  auto alloc = [&](size_t bytes) -> char* {
    char* p = ws + off;
    off = (off + bytes + 255) & ~(size_t)255;
    return p;
  };

  // group 1: padded adjacency (zeroed; padding cols kill K-garbage)
  char* grpA = ws + off;
  h16* Aq16 = (h16*)alloc(NPP * 2);
  h16* Ah16 = (h16*)alloc(NPP * 2);
  h16* Am16 = (h16*)alloc(NPP * 2);
  h16* Aw16 = (h16*)alloc(NPP * 2);          // also serves as A_diff (identical)
  size_t grpA_n = ((ws + off) - grpA) / 2;

  // group 2: padded f16 weights (zeroed; padded K-rows kill G16 garbage)
  char* grpW = ws + off;
  h16* Wg16[4]; h16* Wc16[4];
  const int wgP[4] = { fcat0P * 2 * H_, fcat1 * 2 * H_, fcat0P * 2 * H_, fcat1 * 2 * H_ };
  const int wg_n[4] = { fcat0 * 2 * H_, fcat1 * 2 * H_, fcat0 * 2 * H_, fcat1 * 2 * H_ };
  const int wcP[4] = { fcat0P * H_,     fcat1 * H_,     fcat0P * H_,     fcat1 * H_ };
  const int wc_n[4] = { fcat0 * H_,     fcat1 * H_,     fcat0 * H_,     fcat1 * H_ };
  for (int i = 0; i < 4; ++i) { Wg16[i] = (h16*)alloc((size_t)wgP[i] * 2);
                                Wc16[i] = (h16*)alloc((size_t)wcP[i] * 2); }
  size_t grpW_n = ((ws + off) - grpW) / 2;

  // group 3: f16 activation buffers (zero once for deterministic padding)
  char* grpZ = ws + off;
  h16* Xn16 = (h16*)alloc((size_t)NP_ * (B_ * T_) * 2);
  h16* Z16  = (h16*)alloc((size_t)NP_ * colsP1 * 2);
  h16* Y1   = (h16*)alloc((size_t)NP_ * colsP1 * 2);
  h16* Y2   = (h16*)alloc((size_t)NP_ * colsP1 * 2);
  h16* Y3   = (h16*)alloc((size_t)NP_ * colsP1 * 2);
  h16* Y4   = (h16*)alloc((size_t)NP_ * colsP1 * 2);
  h16* G16  = (h16*)alloc((size_t)B_ * N_ * fcat1 * 2);
  size_t grpZ_n = ((ws + off) - grpZ) / 2;

  float* dall = (float*)alloc(N_ * 4);
  float* dcol = (float*)alloc(N_ * 4);
  float* dbin = (float*)alloc(N_ * 4);
  float* dns  = (float*)alloc(N_ * 4);
  float* Ym   = (float*)alloc((size_t)NP_ * (B_ * T_) * 4);
  float* Yw   = (float*)alloc((size_t)NP_ * (B_ * T_) * 4);
  float* gate = (float*)alloc((size_t)B_ * N_ * E_ * 4);
  float* ximp = (float*)alloc((size_t)T_ * B_ * N_ * 4);
  float* hbuf = (float*)alloc((size_t)2 * B_ * N_ * H_ * 4);
  float* ru   = (float*)alloc((size_t)B_ * N_ * 2 * H_ * 4);
  float* cb   = (float*)alloc((size_t)B_ * N_ * H_ * 4);
  float* dec_in = (float*)alloc((size_t)B_ * N_ * 4);
  (void)ws_size; (void)n_in; (void)in_sizes; (void)out_size;

  // ---- zero padded regions (once per launch; deterministic) ----
  k_zero16<<<4096, 256, 0, stream>>>((h16*)grpA, grpA_n);
  k_zero16<<<1024, 256, 0, stream>>>((h16*)grpW, grpW_n);
  k_zero16<<<4096, 256, 0, stream>>>((h16*)grpZ, grpZ_n);
  k_zero<<<CDIV(2 * B_ * N_ * H_, 256), 256, 0, stream>>>(hbuf, 2 * B_ * N_ * H_);

  // ---- adjacency prep ----
  k_rowstats<<<N_, 256, 0, stream>>>(adj, dall, dbin, dns);
  k_colsum<<<CDIV(N_, 256), 256, 0, stream>>>(adj, dcol);
  k_build_adj<<<CDIV(N_ * N_, 256), 256, 0, stream>>>(adj, dall, dcol, dbin, dns,
                                                      Aq16, Ah16, Am16, Aw16);
  // ---- weights -> f16 (real rows; padded tail stays zero) ----
  for (int i = 0; i < 4; ++i) {
    k_cvt<<<CDIV(wg_n[i], 256), 256, 0, stream>>>(Wg_f32[i], Wg16[i], wg_n[i]);
    k_cvt<<<CDIV(wc_n[i], 256), 256, 0, stream>>>(Wc_f32[i], Wc16[i], wc_n[i]);
  }
  // ---- imputation ----
  k_xn<<<CDIV(N_ * B_ * T_, 256), 256, 0, stream>>>(x_enc, Xn16);
  gemm32(Am16, Xn16, Ym, NP_, B_ * T_, NP_, stream);
  gemm32(Aw16, Xn16, Yw, NP_, B_ * T_, NP_, stream);   // A_diff == A_wmean -> reuse
  k_gate<<<CDIV(B_ * N_, 128), 128, 0, stream>>>(x_enc, gw1, gb1, gw2, gb2, gate);
  k_combine<<<CDIV(B_ * N_, 128), 128, 0, stream>>>(Ym, Yw, ewm, ebm, eww, ebw,
                                                    ewd, ebd, gate, x_enc, ximp);
  // ---- recurrent cells ----
  float* h0 = hbuf;
  float* h1 = hbuf + (size_t)B_ * N_ * H_;

  auto cell = [&](const float* inp, int fin, float* h, int wi) {
    int f = fin + H_;
    int colsP = (fin == 1) ? colsP0 : colsP1;
    int fcat = 5 * f;
    int fcatP = (fin == 1) ? fcat0P : fcat1;
    int nZ = N_ * B_ * f, nG = B_ * N_ * fcat;
    int nru = B_ * N_ * 2 * H_, nc2 = B_ * N_ * H_;
    // gate conv
    k_pack_xh<<<CDIV(nZ, 256), 256, 0, stream>>>(inp, fin, h, nullptr, Z16, colsP);
    gemm16(Aq16, Z16, Y1, NP_, colsP, NP_, stream);
    gemm16(Aq16, Y1,  Y2, NP_, colsP, NP_, stream);
    gemm16(Ah16, Z16, Y3, NP_, colsP, NP_, stream);
    gemm16(Ah16, Y3,  Y4, NP_, colsP, NP_, stream);
    k_pack_cat<<<CDIV(nG, 256), 256, 0, stream>>>(Z16, Y1, Y2, Y3, Y4, G16, f, colsP, fcatP);
    gemm32(G16, Wg16[wi], ru, B_ * N_, 2 * H_, fcatP, stream);
    k_bias_sigmoid<<<CDIV(nru, 256), 256, 0, stream>>>(ru, bg_f32[wi], nru, 2 * H_);
    // candidate conv (uses r*h)
    k_pack_xh<<<CDIV(nZ, 256), 256, 0, stream>>>(inp, fin, h, ru, Z16, colsP);
    gemm16(Aq16, Z16, Y1, NP_, colsP, NP_, stream);
    gemm16(Aq16, Y1,  Y2, NP_, colsP, NP_, stream);
    gemm16(Ah16, Z16, Y3, NP_, colsP, NP_, stream);
    gemm16(Ah16, Y3,  Y4, NP_, colsP, NP_, stream);
    k_pack_cat<<<CDIV(nG, 256), 256, 0, stream>>>(Z16, Y1, Y2, Y3, Y4, G16, f, colsP, fcatP);
    gemm32(G16, Wc16[wi], cb, B_ * N_, H_, fcatP, stream);
    k_bias_tanh<<<CDIV(nc2, 256), 256, 0, stream>>>(cb, bc_f32[wi], nc2, H_);
    k_gru<<<CDIV(nc2, 256), 256, 0, stream>>>(h, ru, cb);
  };

  // encoder: scan over T steps, 2 layers (weight idx 0,1)
  for (int t = 0; t < T_; ++t) {
    cell(ximp + (size_t)t * B_ * N_, 1, h0, 0);
    cell(h0, H_, h1, 1);
  }
  // decoder: autoregressive, OUT_ steps (weight idx 2,3)
  const float* din = ximp + (size_t)(T_ - 1) * B_ * N_;
  for (int t = 0; t < OUT_; ++t) {
    cell(din, 1, h0, 2);
    cell(h0, H_, h1, 3);
    k_fc<<<CDIV(B_ * N_, 256), 256, 0, stream>>>(h1, fcw, fcb, (float*)d_out, dec_in, t);
    din = dec_in;
  }
}
